// DynamicGraphBuilder_18245021073866
// MI455X (gfx1250) — compile-verified, matrix operand order checked by reading, not run on hardware
//
#include <hip/hip_runtime.h>

typedef __attribute__((ext_vector_type(16))) _Float16 v16h;
typedef __attribute__((ext_vector_type(8)))  _Float16 v8h;
typedef __attribute__((ext_vector_type(4)))  _Float16 v4h;
typedef __attribute__((ext_vector_type(8)))  float    v8f;

#define NN   19      // nodes
#define DIM  128     // feature dim
#define TQ   2048    // time steps
#define PITCH 20     // LDS row pitch for 19x19 grids

__global__ __launch_bounds__(32)
void dyn_graph_kernel(const float* __restrict__ feat, float* __restrict__ out) {
    // 32 rows x 128 halves of normalized features (rows 19..31 zeroed) = 8 KB
    __shared__ __align__(16) _Float16 Ash[32 * DIM];
    __shared__ float Glds[NN * PITCH];   // logits (sim / temperature)
    __shared__ float Adj[NN * PITCH];    // sparsified softmax

    const int lane = threadIdx.x;        // 0..31 (wave32)
    const int bt   = blockIdx.x;         // b*2048 + t
    const int b    = bt >> 11;
    const int t    = bt & (TQ - 1);

    // ---------------- Phase 1: stream rows, normalize, stage f16 in LDS ------
    const float4* fp = (const float4*)feat;
    const size_t rowbase = ((size_t)(b * NN) * TQ + t) * (DIM / 4);

    float4 rows[NN];
#pragma unroll
    for (int n = 0; n < NN; ++n)
        rows[n] = fp[rowbase + (size_t)n * (TQ * (DIM / 4)) + lane];

#pragma unroll
    for (int n = 0; n < NN; ++n) {
        float4 v = rows[n];
        float ss = v.x * v.x + v.y * v.y + v.z * v.z + v.w * v.w;
#pragma unroll
        for (int off = 16; off > 0; off >>= 1)
            ss += __shfl_xor(ss, off, 32);
        float inv = 1.0f / fmaxf(sqrtf(ss), 1e-12f);   // matches reference eps
        v4h h;
        h[0] = (_Float16)(v.x * inv);
        h[1] = (_Float16)(v.y * inv);
        h[2] = (_Float16)(v.z * inv);
        h[3] = (_Float16)(v.w * inv);
        *(v4h*)(&Ash[n * DIM + lane * 4]) = h;
    }
    // zero pad rows 19..31 so padded Gram entries are well-defined
#pragma unroll
    for (int n = NN; n < 32; ++n) {
        v4h z = {};
        *(v4h*)(&Ash[n * DIM + lane * 4]) = z;
    }
    __syncthreads();

    // ---------------- Phase 2: Gram = A * A^T via WMMA -----------------------
    // A fragment (16x32 f16, row-tile mi) doubles as the B fragment for the
    // symmetric product (mirrored ISA layouts hold identical data per lane).
    v8f acc00 = {}, acc01 = {}, acc10 = {}, acc11 = {};
    const int grp = lane >> 4;     // 0 or 1 (K-half group)
    const int lm  = lane & 15;     // row within 16-tile

#pragma unroll
    for (int kb = 0; kb < 4; ++kb) {
        const int base = kb * 32 + grp * 8;            // half index within row
        v8h lo0 = *(v8h*)(&Ash[lm        * DIM + base]);
        v8h hi0 = *(v8h*)(&Ash[lm        * DIM + base + 16]);
        v8h lo1 = *(v8h*)(&Ash[(lm + 16) * DIM + base]);
        v8h hi1 = *(v8h*)(&Ash[(lm + 16) * DIM + base + 16]);
        v16h f0, f1;
#pragma unroll
        for (int i = 0; i < 8; ++i) {
            f0[i] = lo0[i]; f0[i + 8] = hi0[i];
            f1[i] = lo1[i]; f1[i + 8] = hi1[i];
        }
        acc00 = __builtin_amdgcn_wmma_f32_16x16x32_f16(false, f0, false, f0, (short)0, acc00, false, false);
        acc01 = __builtin_amdgcn_wmma_f32_16x16x32_f16(false, f0, false, f1, (short)0, acc01, false, false);
        acc10 = __builtin_amdgcn_wmma_f32_16x16x32_f16(false, f1, false, f0, (short)0, acc10, false, false);
        acc11 = __builtin_amdgcn_wmma_f32_16x16x32_f16(false, f1, false, f1, (short)0, acc11, false, false);
    }

    // ---------------- Phase 3: scatter logits = sim / 0.1 to LDS -------------
    // C/D layout: VGPR r, lane L -> row = tile_m*16 + (L>>4)*8 + r, col = tile_n*16 + (L&15)
#pragma unroll
    for (int r = 0; r < 8; ++r) {
        const int row0 = grp * 8 + r;          // 0..15, always < 19
        const int row1 = 16 + grp * 8 + r;     // 16..31
        Glds[row0 * PITCH + lm] = acc00[r] * 10.0f;
        if (lm + 16 < NN) Glds[row0 * PITCH + lm + 16] = acc01[r] * 10.0f;
        if (row1 < NN) {
            Glds[row1 * PITCH + lm] = acc10[r] * 10.0f;
            if (lm + 16 < NN) Glds[row1 * PITCH + lm + 16] = acc11[r] * 10.0f;
        }
    }
    __syncthreads();

    // ---------------- Phase 4: per-row softmax + top-3 + threshold -----------
    if (lane < NN) {
        float x[NN];
        float m = -1e30f;
#pragma unroll
        for (int j = 0; j < NN; ++j) {
            x[j] = Glds[lane * PITCH + j];
            m = fmaxf(m, x[j]);
        }
        float s = 0.0f;
#pragma unroll
        for (int j = 0; j < NN; ++j) s += __expf(x[j] - m);
        // 3rd largest logit (softmax is monotone, so top-3 of logits == top-3 of probs)
        float t1 = -1e30f, t2 = -1e30f, t3 = -1e30f;
#pragma unroll
        for (int j = 0; j < NN; ++j) {
            float v = x[j];
            if (v > t1)      { t3 = t2; t2 = t1; t1 = v; }
            else if (v > t2) { t3 = t2; t2 = v; }
            else if (v > t3) { t3 = v; }
        }
        const float invs = 1.0f / s;
#pragma unroll
        for (int j = 0; j < NN; ++j) {
            float p = __expf(x[j] - m) * invs;
            p = (x[j] >= t3) ? p : 0.0f;     // top-k sparsify
            p = (p > 1e-4f) ? p : 0.0f;      // threshold prune
            Adj[lane * PITCH + j] = p;
        }
    }
    __syncthreads();

    // ---------------- Phase 5: symmetrize + store ----------------------------
    float* outp = out + (size_t)bt * (NN * NN);
#pragma unroll
    for (int idx = lane; idx < NN * NN; idx += 32) {
        const int i = idx / NN;
        const int j = idx - i * NN;
        outp[idx] = 0.5f * (Adj[i * PITCH + j] + Adj[j * PITCH + i]);
    }
}

extern "C" void kernel_launch(void* const* d_in, const int* in_sizes, int n_in,
                              void* d_out, int out_size, void* d_ws, size_t ws_size,
                              hipStream_t stream) {
    const float* feat = (const float*)d_in[0];
    float* out = (float*)d_out;
    (void)in_sizes; (void)n_in; (void)out_size; (void)d_ws; (void)ws_size;

    const int sites = 32 * TQ;                  // B * T = 65536 (b,t) sites
    dyn_graph_kernel<<<dim3(sites), dim3(32), 0, stream>>>(feat, out);
}
